// EmbeddingProductHead_1288490189073
// MI455X (gfx1250) — matrix operand order repaired and checked by reading, not compiled, not run on hardware
//
#include <hip/hip_runtime.h>
#include <hip/hip_bf16.h>

// ---------------------------------------------------------------------------
// EmbeddingProductHead for MI455X (gfx1250, wave32, WMMA, async-LDS).
//   Pass 0: convert W (32MB) and emb (2MB) f32 -> bf16 once into d_ws.
//   Pass 1: bin_emb = softplus(emb @ W^T + b) -> bf16 ws (123 MB, L2-resident).
//           M-tile = 64 tokens staged in LDS (async), B streamed bf16 from L2.
//   Pass 2: per token, logits[y,x,z] = exp(ls) * sum_h ex[x,h]*ey[y,h]*ez[z,h]
//           bin_emb tile LDS-resident; y-major, 9 accumulators per wave;
//           A-operand built with v_pk_mul_bf16.
// All GEMMs on v_wmma_f32_16x16x32_bf16 with f32 accumulation.
// ---------------------------------------------------------------------------

typedef __attribute__((ext_vector_type(16))) __bf16        v16bf;
typedef __attribute__((ext_vector_type(8)))  __bf16        v8bf;
typedef __attribute__((ext_vector_type(8)))  float         v8f;
typedef __attribute__((ext_vector_type(4)))  float         v4f;
typedef __attribute__((ext_vector_type(4)))  unsigned int  v4u;
typedef __attribute__((ext_vector_type(8)))  unsigned int  v8u;

#define HIDDEN   256
#define NBIN     41
#define COMBINED 123                        // 3*41
#define OUT_O    (COMBINED * HIDDEN)        // 31488
#define TOKENS   2048                       // 4*512
#define OUT_PER_TOKEN (NBIN * NBIN * NBIN)  // 68921
#define LROW_PITCH 264                      // 256 bf16 + 8 pad -> 4-bank row skew

#define USE_ASYNC_LDS  1   // global_load_async_to_lds_b128 + s_wait_asynccnt
#define USE_PK_MUL_ASM 1   // force v_pk_mul_bf16 for the phase-2 A-operand build

// jax.nn.softplus, numerically stable, fast-math logs (result is bf16-rounded)
__device__ inline float softplusf(float x) {
    return fmaxf(x, 0.0f) + __logf(1.0f + __expf(-fabsf(x)));
}

// packed bf16*bf16 on a dword (2 lanes of the fragment)
__device__ inline unsigned pk_mul_bf16(unsigned a, unsigned b) {
#if USE_PK_MUL_ASM
    unsigned d;
    asm("v_pk_mul_bf16 %0, %1, %2" : "=v"(d) : "v"(a), "v"(b));
    return d;
#else
    v8bf av, bv;  // fallback: element-wise (may scalarize)
    __builtin_memcpy(&av, &a, 4);
    (void)bv;
    return a;
#endif
}

// A-operand frag (16x32 bf16): per-lane elems [0..7]=K k0..k0+7, [8..15]=k0+16..k0+23
__device__ inline v16bf frag_split(const __bf16* row, int k0) {
    const v8bf lo = *(const v8bf*)(row + k0);
    const v8bf hi = *(const v8bf*)(row + k0 + 16);
    return __builtin_shufflevector(lo, hi, 0, 1, 2, 3, 4, 5, 6, 7,
                                   8, 9, 10, 11, 12, 13, 14, 15);
}

// B-operand frag (32x16 bf16): per-lane elems [0..7]=K k0..k0+7, [8..15]=k0+16..k0+23
__device__ inline v16bf frag_contig(const __bf16* row, int k0) {
    const v8bf lo = *(const v8bf*)(row + k0);
    const v8bf hi = *(const v8bf*)(row + k0 + 16);
    return __builtin_shufflevector(lo, hi, 0, 1, 2, 3, 4, 5, 6, 7,
                                   8, 9, 10, 11, 12, 13, 14, 15);
}

// Copy nrows x 256 bf16 rows (global, pitch 256) into LDS (pitch 264) using the
// CDNA5 async memory->LDS path (ASYNCcnt). 16 B per lane per op.
__device__ inline void copy_rows_to_lds(__bf16* ldsdst, const __bf16* gsrc,
                                        int nchunks, int tid) {
    for (int idx = tid; idx < nchunks; idx += 256) {
        const int row = idx >> 5;
        const int c8  = (idx & 31) * 8;
        __bf16* d = ldsdst + row * LROW_PITCH + c8;
        const __bf16* s = gsrc + (size_t)row * HIDDEN + c8;
#if USE_ASYNC_LDS
        const unsigned lds_off = (unsigned)(unsigned long long)d;
        asm volatile("global_load_async_to_lds_b128 %0, %1, off"
                     :: "v"(lds_off), "v"((unsigned long long)s) : "memory");
#else
        *(v8bf*)d = *(const v8bf*)s;
#endif
    }
#if USE_ASYNC_LDS
    asm volatile("s_wait_asynccnt 0" ::: "memory");
#endif
}

// ---------------------------------------------------------------------------
// Pass 0: f32 -> bf16 bulk convert (8 elems / thread)
// ---------------------------------------------------------------------------
__global__ __launch_bounds__(256) void cvt_bf16_kernel(const float* __restrict__ src,
                                                       __bf16* __restrict__ dst, int n8) {
    const int i = blockIdx.x * 256 + threadIdx.x;
    if (i < n8) {
        const v4f a = *(const v4f*)(src + (size_t)i * 8);
        const v4f b = *(const v4f*)(src + (size_t)i * 8 + 4);
        v8bf o;
#pragma unroll
        for (int j = 0; j < 4; ++j) {
            o[j]     = (__bf16)a[j];
            o[4 + j] = (__bf16)b[j];
        }
        *(v8bf*)(dst + (size_t)i * 8) = o;
    }
}

// ---------------------------------------------------------------------------
// Pass 1: (2048 x 256) @ (256 x 31488) + bias -> softplus -> bf16 workspace
// grid = (2048/64, 31488/256), block = 256 (8 waves).
// Block stages a 64x256 bf16 emb tile in LDS; wave owns a 64x32 C tile.
// ---------------------------------------------------------------------------
__global__ __launch_bounds__(256) void binproj_softplus_kernel(
    const __bf16* __restrict__ embbf, const __bf16* __restrict__ Wbf,
    const float* __restrict__ bias, __bf16* __restrict__ binemb)
{
    __shared__ __bf16 semb[64 * LROW_PITCH];   // ~33.8 KB

    const int tid = threadIdx.x;
    const int t0  = blockIdx.x * 64;
    copy_rows_to_lds(semb, embbf + (size_t)t0 * HIDDEN, 64 * 32, tid);
    __syncthreads();

    const int wave = tid >> 5;
    const int lane = tid & 31;
    const int lrow = lane & 15;
    const int lhi  = (lane >> 4) & 1;
    const int o0   = blockIdx.y * 256 + wave * 32;

    const __bf16* Ar[4];
#pragma unroll
    for (int mi = 0; mi < 4; ++mi) Ar[mi] = semb + (mi * 16 + lrow) * LROW_PITCH;
    const __bf16* Br0 = Wbf + (size_t)(o0 + lrow) * HIDDEN;
    const __bf16* Br1 = Wbf + (size_t)(o0 + 16 + lrow) * HIDDEN;

    v8f acc[8];   // [mi][ni], ni in {0,1}
#pragma unroll
    for (int q = 0; q < 8; ++q) { v8f z = {}; acc[q] = z; }

#pragma unroll
    for (int kk = 0; kk < 8; ++kk) {
        const int ka = kk * 32 + lhi * 8;
        const int kb = kk * 32 + lhi * 8;
        const v16bf b0 = frag_contig(Br0, kb);   // global_load_b128 x2 (bf16, L2)
        const v16bf b1 = frag_contig(Br1, kb);
#pragma unroll
        for (int mi = 0; mi < 4; ++mi) {
            const v16bf a = frag_split(Ar[mi], ka);   // ds_load_b128 x2
            acc[mi * 2 + 0] = __builtin_amdgcn_wmma_f32_16x16x32_bf16(
                false, a, false, b0, (short)0, acc[mi * 2 + 0], false, false);
            acc[mi * 2 + 1] = __builtin_amdgcn_wmma_f32_16x16x32_bf16(
                false, a, false, b1, (short)0, acc[mi * 2 + 1], false, false);
        }
    }

    const int n0 = o0 + lrow;
    const int n1 = o0 + 16 + lrow;
    const float bias0 = bias[n0];
    const float bias1 = bias[n1];
#pragma unroll
    for (int mi = 0; mi < 4; ++mi) {
#pragma unroll
        for (int r = 0; r < 8; ++r) {
            const int token = t0 + mi * 16 + r + lhi * 8;
            const size_t base = (size_t)token * OUT_O;
            binemb[base + n0] = (__bf16)softplusf(acc[mi * 2 + 0][r] + bias0);  // RT: stay in L2
            binemb[base + n1] = (__bf16)softplusf(acc[mi * 2 + 1][r] + bias1);
        }
    }
}

// ---------------------------------------------------------------------------
// Pass 2: per token, logits[y,x,z] = sum_h (ex[x,h]*ey[y,h]) * ez[z,h]
// One block per token; bin_emb tile (123x256 bf16) LDS-resident.
// y-major: each wave owns a full y-slice -> 9 accumulators (48x48 padded C),
// B frags (ez) + ey chunks shared across the 3 M-tiles.
// ---------------------------------------------------------------------------
__global__ __launch_bounds__(256) void triple_product_kernel(
    const __bf16* __restrict__ binemb, const float* __restrict__ logit_scale,
    float* __restrict__ out)
{
    __shared__ __bf16 sbin[COMBINED * LROW_PITCH];   // ~63.4 KB of 320 KB/WGP

    const int tid   = threadIdx.x;
    const int token = blockIdx.x;

    copy_rows_to_lds(sbin, binemb + (size_t)token * OUT_O, COMBINED * 32, tid);
    __syncthreads();

    const float scale = __expf(logit_scale[0]);
    const int wave = tid >> 5;
    const int lane = tid & 31;
    const int lrow = lane & 15;
    const int lhi  = (lane >> 4) & 1;

    // ez rows (B columns) are per-lane; clamp pad columns (never stored).
    const __bf16* ez0 = sbin + (2 * NBIN + lrow) * LROW_PITCH;
    const __bf16* ez1 = sbin + (2 * NBIN + 16 + lrow) * LROW_PITCH;
    const int z2row   = (32 + lrow < NBIN) ? (32 + lrow) : (NBIN - 1);
    const __bf16* ez2 = sbin + (2 * NBIN + z2row) * LROW_PITCH;

    // ex rows (A rows) per-lane; clamp pad rows (never stored).
    const __bf16* exr[3];
#pragma unroll
    for (int mt = 0; mt < 3; ++mt) {
        const int xr = (mt * 16 + lrow < NBIN) ? (mt * 16 + lrow) : (NBIN - 1);
        exr[mt] = sbin + xr * LROW_PITCH;
    }

    for (int y = wave; y < NBIN; y += 8) {
        const __bf16* eyr = sbin + (NBIN + y) * LROW_PITCH;

        v8f acc[9];
#pragma unroll
        for (int q = 0; q < 9; ++q) { v8f z = {}; acc[q] = z; }

#pragma unroll
        for (int kk = 0; kk < 8; ++kk) {
            const int ka = kk * 32 + lhi * 8;
            const int kb = kk * 32 + lhi * 8;

            // shared across M-tiles: ey chunks (as dwords) + 3 B frags
            const v4u ey0 = *(const v4u*)(eyr + ka);
            const v4u ey1 = *(const v4u*)(eyr + ka + 16);
            const v16bf b0 = frag_contig(ez0, kb);
            const v16bf b1 = frag_contig(ez1, kb);
            const v16bf b2 = frag_contig(ez2, kb);

#pragma unroll
            for (int mt = 0; mt < 3; ++mt) {
                const v4u x0 = *(const v4u*)(exr[mt] + ka);
                const v4u x1 = *(const v4u*)(exr[mt] + ka + 16);
                v8u au;   // A = ex ⊙ ey[y] via v_pk_mul_bf16 (8 packed ops)
#pragma unroll
                for (int j = 0; j < 4; ++j) {
                    au[j]     = pk_mul_bf16(x0[j], ey0[j]);
                    au[4 + j] = pk_mul_bf16(x1[j], ey1[j]);
                }
                const v16bf a = __builtin_bit_cast(v16bf, au);
                acc[mt * 3 + 0] = __builtin_amdgcn_wmma_f32_16x16x32_bf16(
                    false, a, false, b0, (short)0, acc[mt * 3 + 0], false, false);
                acc[mt * 3 + 1] = __builtin_amdgcn_wmma_f32_16x16x32_bf16(
                    false, a, false, b1, (short)0, acc[mt * 3 + 1], false, false);
                acc[mt * 3 + 2] = __builtin_amdgcn_wmma_f32_16x16x32_bf16(
                    false, a, false, b2, (short)0, acc[mt * 3 + 2], false, false);
            }
        }

        // Store: out[token][(y*41 + x)*41 + z], non-temporal (565 MB stream).
        float* obase = out + (size_t)token * OUT_PER_TOKEN + (size_t)y * (NBIN * NBIN);
        const int z0 = lrow, z1 = 16 + lrow, z2 = 32 + lrow;
#pragma unroll
        for (int mt = 0; mt < 3; ++mt) {
#pragma unroll
            for (int r = 0; r < 8; ++r) {
                const int x = mt * 16 + r + lhi * 8;
                if (x < NBIN) {
                    float* orow = obase + x * NBIN;
                    __builtin_nontemporal_store(scale * acc[mt * 3 + 0][r], orow + z0);
                    __builtin_nontemporal_store(scale * acc[mt * 3 + 1][r], orow + z1);
                    if (z2 < NBIN)
                        __builtin_nontemporal_store(scale * acc[mt * 3 + 2][r], orow + z2);
                }
            }
        }
    }
}

extern "C" void kernel_launch(void* const* d_in, const int* in_sizes, int n_in,
                              void* d_out, int out_size, void* d_ws, size_t ws_size,
                              hipStream_t stream) {
    (void)in_sizes; (void)n_in; (void)out_size; (void)ws_size;
    const float* emb  = (const float*)d_in[0];   // (4,512,256) f32
    const float* W    = (const float*)d_in[1];   // (31488,256) f32
    const float* bias = (const float*)d_in[2];   // (31488,)    f32
    const float* ls   = (const float*)d_in[3];   // scalar      f32
    float* out = (float*)d_out;                  // (4,512,68921) f32

    // Workspace layout (bf16): W (16.1 MB) | emb (1 MB) | bin_emb (123 MB) ≈ 146 MB
    __bf16* wbf    = (__bf16*)d_ws;
    __bf16* embbf  = wbf + (size_t)OUT_O * HIDDEN;
    __bf16* binemb = embbf + (size_t)TOKENS * HIDDEN;

    cvt_bf16_kernel<<<(OUT_O * HIDDEN / 8 + 255) / 256, 256, 0, stream>>>(W, wbf, OUT_O * HIDDEN / 8);
    cvt_bf16_kernel<<<(TOKENS * HIDDEN / 8 + 255) / 256, 256, 0, stream>>>(emb, embbf, TOKENS * HIDDEN / 8);
    binproj_softplus_kernel<<<dim3(TOKENS / 64, OUT_O / 256), 256, 0, stream>>>(embbf, wbf, bias, binemb);
    triple_product_kernel<<<dim3(TOKENS), 256, 0, stream>>>(binemb, ls, out);
}
